// OptimizedContextualizer_84679575208503
// MI455X (gfx1250) — matrix operand order repaired and checked by reading, not compiled, not run on hardware
//
#include <hip/hip_runtime.h>
#include <math.h>

typedef __attribute__((ext_vector_type(16))) _Float16 v16h;
typedef __attribute__((ext_vector_type(8)))  _Float16 v8h;
typedef __attribute__((ext_vector_type(8)))  float    v8f;
typedef __attribute__((ext_vector_type(4)))  float    v4f;

#define B_DIM 4
#define T_DIM 4096
#define DH    512     // D/2, the normalized/gated half
#define DFULL 1024

// ---------------------------------------------------------------------------
// Kernel 1: xn = x_right / max(||x_right||, 1e-12), stored as f16 (row major).
// One wave per row; each lane owns 16 contiguous d-elements.
// ---------------------------------------------------------------------------
__global__ __launch_bounds__(256)
void xnorm_f16_kernel(const float* __restrict__ x, _Float16* __restrict__ xnh) {
  const int row  = blockIdx.x * 8 + (threadIdx.x >> 5);
  const int lane = threadIdx.x & 31;
  const float* src = x + (size_t)row * DFULL + DH + lane * 16;
  float v[16];
  float s = 0.f;
#pragma unroll
  for (int i = 0; i < 4; ++i) {
    v4f t = *(const v4f*)(src + 4 * i);
#pragma unroll
    for (int e = 0; e < 4; ++e) { v[4 * i + e] = t[e]; s += t[e] * t[e]; }
  }
#pragma unroll
  for (int off = 16; off >= 1; off >>= 1) s += __shfl_xor(s, off, 32);
  const float scale = 1.0f / fmaxf(sqrtf(s), 1e-12f);

  v8h h0, h1;
#pragma unroll
  for (int e = 0; e < 8; ++e) {
    h0[e] = (_Float16)(v[e] * scale);
    h1[e] = (_Float16)(v[8 + e] * scale);
  }
  _Float16* dst = xnh + (size_t)row * DH + lane * 16;
  *(v8h*)(dst)     = h0;
  *(v8h*)(dst + 8) = h1;
}

// ---------------------------------------------------------------------------
// Kernel 2: one-time tiled transpose xnh[b][t][d] -> xnT[b][d][t].
// 64x64 f16 tiles through LDS; vectorized global load + store.
// ---------------------------------------------------------------------------
__global__ __launch_bounds__(256)
void xtrans_kernel(const _Float16* __restrict__ xnh, _Float16* __restrict__ xnT) {
  __shared__ __attribute__((aligned(16))) _Float16 tile[64 * 72];  // padded rows
  const int b  = blockIdx.z;
  const int t0 = blockIdx.x * 64;
  const int d0 = blockIdx.y * 64;
  const int r  = threadIdx.x >> 2;   // 0..63
  const int s  = threadIdx.x & 3;    // 0..3 : 16-half segment

  const _Float16* src = xnh + ((size_t)b * T_DIM + t0 + r) * DH + d0 + s * 16;
#pragma unroll
  for (int i = 0; i < 2; ++i)
    *(v8h*)(tile + r * 72 + s * 16 + i * 8) = *(const v8h*)(src + i * 8);
  __syncthreads();

  // transposed gather: output row = d, cols = t
  _Float16* dst = xnT + ((size_t)b * DH + d0 + r) * T_DIM + t0 + s * 16;
  v8h g0, g1;
#pragma unroll
  for (int e = 0; e < 8; ++e) {
    g0[e] = tile[(s * 16 + e) * 72 + r];
    g1[e] = tile[(s * 16 + 8 + e) * 72 + r];
  }
  *(v8h*)(dst)     = g0;
  *(v8h*)(dst + 8) = g1;
}

// ---------------------------------------------------------------------------
// Kernel 3: fused  sim = xn xn^T ; w = V*sim ; ctx = w xn ;
//           out = x_left * sigmoid(ctx + bias)
// Workgroup: 256 threads = 8 waves, owns 32 query rows. Keys in blocks of 64.
// GEMM1 B reads xnh (d-contiguous); GEMM2 B reads xnT (k-contiguous).
// ---------------------------------------------------------------------------
__global__ __launch_bounds__(256)
void ctx_kernel(const float* __restrict__ x, const float* __restrict__ V,
                const float* __restrict__ bias,
                const _Float16* __restrict__ xnh, const _Float16* __restrict__ xnT,
                float* __restrict__ out) {
  __shared__ __attribute__((aligned(32))) _Float16 xq[32 * DH];  // 32 KB [q][d]
  __shared__ __attribute__((aligned(32))) _Float16 Wt[32 * 64];  //  4 KB [q][k]

  const int b    = blockIdx.y;
  const int q0   = blockIdx.x * 32;
  const int tid  = threadIdx.x;
  const int wv   = tid >> 5;
  const int lane = tid & 31;
  const int ln16 = lane & 15;
  const int hi8  = (lane >= 16) ? 8 : 0;    // C-layout M offset
  const int hi16 = (lane >= 16) ? 16 : 0;   // A/B-layout K base

  const int qi = wv >> 2, ki = wv & 3;      // GEMM1 tile of this wave
  const int h  = wv >> 2;                   // GEMM2 q-half
  const int db = (wv & 3) * 128;            // GEMM2 d-slice base

  // stage query tile (32 x 512 f16) to LDS, once
  {
    const int q = tid >> 3, seg = tid & 7;
    const _Float16* src = xnh + ((size_t)b * T_DIM + q0 + q) * DH + seg * 64;
    _Float16* dst = xq + q * DH + seg * 64;
#pragma unroll
    for (int i = 0; i < 8; ++i) *(v8h*)(dst + i * 8) = *(const v8h*)(src + i * 8);
  }
  __syncthreads();

  v8f acc[8];
#pragma unroll
  for (int i = 0; i < 8; ++i) acc[i] = (v8f){};

  const _Float16* xnT_b = xnT + (size_t)b * DH * T_DIM;

  for (int k0 = 0; k0 < T_DIM; k0 += 64) {
    // ---- GEMM1: sim(16x16) = xq(16x512) . xk^T, one tile per wave ------
    v8f sim = (v8f){};
    {
      const _Float16* xqrow = xq + (qi * 16 + ln16) * DH;                 // A: M=q
      const _Float16* xkrow =                                            // B: N=key
          xnh + ((size_t)b * T_DIM + k0 + ki * 16 + ln16) * DH + hi16;
      if (k0 + 64 < T_DIM)  // speculative prefetch of next key block row
        __builtin_prefetch(xkrow + (size_t)64 * DH, 0, 3);
#pragma unroll
      for (int c = 0; c < 16; ++c) {
        v8h alo = *(const v8h*)(xqrow + c * 32 + hi16);
        v8h ahi = *(const v8h*)(xqrow + c * 32 + hi16 + 8);
        v16h a;
#pragma unroll
        for (int i = 0; i < 8; ++i) { a[i] = alo[i]; a[8 + i] = ahi[i]; }
        v16h bf = *(const v16h*)(xkrow + c * 32);
        sim = __builtin_amdgcn_wmma_f32_16x16x32_f16(false, a, false, bf,
                                                     (short)0, sim, false, false);
      }
    }

    // ---- weighted = V * sim, to f16 in LDS (C layout -> row major) -----
    {
      const float* vp = V + (size_t)(q0 + qi * 16 + hi8) * T_DIM + (k0 + ki * 16 + ln16);
      _Float16* wp = Wt + (qi * 16 + hi8) * 64 + ki * 16 + ln16;
#pragma unroll
      for (int j = 0; j < 8; ++j)
        wp[j * 64] = (_Float16)(sim[j] * vp[(size_t)j * T_DIM]);
    }
    __syncthreads();

    // ---- GEMM2: ctx(16 x 128-slice) += Wt(16x64) . xn(64x512) ----------
#pragma unroll
    for (int kc = 0; kc < 2; ++kc) {
      const _Float16* wrow = Wt + (h * 16 + ln16) * 64 + kc * 32 + hi16;
      v8h alo = *(const v8h*)(wrow);
      v8h ahi = *(const v8h*)(wrow + 8);
      v16h a;
#pragma unroll
      for (int i = 0; i < 8; ++i) { a[i] = alo[i]; a[8 + i] = ahi[i]; }
#pragma unroll
      for (int dt = 0; dt < 8; ++dt) {
        const int d = db + dt * 16 + ln16;
        v16h bf = *(const v16h*)(xnT_b + (size_t)d * T_DIM + k0 + kc * 32 + hi16);
        acc[dt] = __builtin_amdgcn_wmma_f32_16x16x32_f16(false, a, false, bf,
                                                         (short)0, acc[dt], false, false);
      }
    }
    __syncthreads();   // protect Wt for next iteration
  }

  // ---- epilogue: out = x_left * sigmoid(ctx + bias), non-temporal ------
  // sigmoid via hardware v_exp_f32 (__expf): gate precision >> f16 GEMM noise
#pragma unroll
  for (int dt = 0; dt < 8; ++dt) {
    const int d = db + dt * 16 + ln16;
    const float bv = bias[d];
#pragma unroll
    for (int j = 0; j < 8; ++j) {
      const int q = q0 + h * 16 + hi8 + j;
      const size_t ro = (size_t)b * T_DIM + q;
      const float xl = __builtin_nontemporal_load(&x[ro * DFULL + d]);
      const float g  = 1.0f / (1.0f + __expf(-(acc[dt][j] + bv)));
      __builtin_nontemporal_store(xl * g, &out[ro * DH + d]);
    }
  }
}

// ---------------------------------------------------------------------------
extern "C" void kernel_launch(void* const* d_in, const int* in_sizes, int n_in,
                              void* d_out, int out_size, void* d_ws, size_t ws_size,
                              hipStream_t stream) {
  const float* x    = (const float*)d_in[0];   // (4, 4096, 1024)
  const float* V    = (const float*)d_in[1];   // (4096, 4096)
  const float* bias = (const float*)d_in[2];   // (512)
  float* out        = (float*)d_out;           // (4, 4096, 512)

  _Float16* xnh = (_Float16*)d_ws;                              // 16 MB
  _Float16* xnT = xnh + (size_t)B_DIM * T_DIM * DH;             // 16 MB

  // Pass 1: normalize right half -> f16 row-major
  dim3 g1(B_DIM * T_DIM / 8);
  xnorm_f16_kernel<<<g1, 256, 0, stream>>>(x, xnh);

  // Pass 2: one-time transpose -> xnT[b][d][t]
  dim3 g2(T_DIM / 64, DH / 64, B_DIM);
  xtrans_kernel<<<g2, 256, 0, stream>>>(xnh, xnT);

  // Pass 3: fused dual-GEMM + gate
  dim3 g3(T_DIM / 32, B_DIM);
  ctx_kernel<<<g3, 256, 0, stream>>>(x, V, bias, xnh, xnT, out);
}
